// SoftCVQLayer_47270410059802
// MI455X (gfx1250) — matrix operand order repaired and checked by reading, not compiled, not run on hardware
//
#include <hip/hip_runtime.h>
#include <hip/hip_bf16.h>

// ---------------------------------------------------------------------------
// SoftCVQ layer for MI455X (gfx1250), wave32 + v_wmma_f32_16x16x32_f16
// + Tensor Data Mover staging of GEMM weight tiles into LDS
// ---------------------------------------------------------------------------

#define BB   8
#define TT   2048
#define DD   768
#define VQ   64
#define KCB  4096      // codebook entries
#define HID  1024
#define L2B  12
#define NMID 4
#define ROWS_BT (BB*TT)   // 16384

typedef __attribute__((ext_vector_type(16))) _Float16 v16h;
typedef __attribute__((ext_vector_type(8)))  float    v8f;
typedef __attribute__((ext_vector_type(8)))  _Float16 h8;

union Frag16 { v16h v; h8 p[2]; _Float16 h[16]; };
union FragC  { v8f v;  float f[8]; };

__device__ __forceinline__ v8f wmma_f16(v16h a, v16h b, v8f c) {
    // D = A(16x32 f16) * B(32x16 f16) + C(16x16 f32)
    return __builtin_amdgcn_wmma_f32_16x16x32_f16(
        /*neg_a=*/false, a, /*neg_b=*/false, b,
        /*c_mod=*/(short)0, c, /*reuse_a=*/false, /*reuse_b=*/false);
}

#if defined(__has_builtin)
#  if __has_builtin(__builtin_amdgcn_tensor_load_to_lds)
#    define HAVE_TDM 1
#  else
#    define HAVE_TDM 0
#  endif
#else
#  define HAVE_TDM 0
#endif

#if HAVE_TDM
// Issue one TDM descriptor: copy a 64-row x KC-half tile of a row-major f16
// matrix (row stride HID halves) from global memory into LDS (rows packed
// contiguously, KC halves each).  D# layout per CDNA5 ISA ch.8.
#define KC 128   // halves per staged K-chunk (64 rows x 128 halves = 16 KB)
__device__ __forceinline__ void tdm_load_tile(const _Float16* gsrc, unsigned lds_byte_off) {
    typedef unsigned int u32x4 __attribute__((ext_vector_type(4)));
    typedef int          i32x4 __attribute__((ext_vector_type(4)));
    typedef int          i32x8 __attribute__((ext_vector_type(8)));
    unsigned long long ga = (unsigned long long)gsrc;
    u32x4 g0;
    g0.x = 1u;                                               // count=1, user D#
    g0.y = lds_byte_off;                                     // lds_addr [63:32]
    g0.z = (unsigned)(ga & 0xffffffffu);                     // global_addr[31:0]
    g0.w = (unsigned)((ga >> 32) & 0x1ffffffu) | (2u << 30); // addr[56:32] | type=2
    i32x8 g1;
    g1[0] = 3 << 16;                  // data_size = 8 bytes (code 3)
    g1[1] = (int)(256u << 16);        // tensor_dim0 = 256 (8B units) -> bits 63:48
    g1[2] = (int)(1024u << 16);       // tensor_dim1 = 1024 rows      -> bits 95:80
    g1[3] = (int)((KC / 4) << 16);    // tile_dim0 = KC/4 (8B units)  -> bits 127:112
    g1[4] = 64;                       // tile_dim1 = 64 rows          -> bits 143:128
    g1[5] = 256;                      // tensor_dim0_stride = 256 (8B units)
    g1[6] = 0;
    g1[7] = 0;
    i32x4 z4 = {0, 0, 0, 0};
#if __clang_major__ >= 23
    i32x8 z8 = {0, 0, 0, 0, 0, 0, 0, 0};
    __builtin_amdgcn_tensor_load_to_lds(g0, g1, z4, z4, z8, 0);
#else
    __builtin_amdgcn_tensor_load_to_lds(g0, g1, z4, z4, 0);
#endif
}
#endif // HAVE_TDM

// ---------------------------------------------------------------------------
// f32 -> f16 convert (weights)
// ---------------------------------------------------------------------------
__global__ void cvt_f16_kernel(const float* __restrict__ S, _Float16* __restrict__ Dst, int n) {
    int i = blockIdx.x * blockDim.x + threadIdx.x;
    if (i < n) Dst[i] = (_Float16)S[i];
}

// ---------------------------------------------------------------------------
// Codebook layer 0: y[m][n] = b_in[n] + sum_k bit(m,k)*w_in[n][k]
// bits[m][k] = (m >> (11-k)) & 1
// ---------------------------------------------------------------------------
__global__ void codebook_l0_kernel(const float* __restrict__ Win, const float* __restrict__ bin,
                                   float* __restrict__ Y) {
    int i = blockIdx.x * blockDim.x + threadIdx.x;   // over KCB*HID
    int m = i >> 10, n = i & (HID - 1);
    const float* w = Win + n * L2B;
    float acc = bin[n];
#pragma unroll
    for (int k = 0; k < L2B; ++k)
        if ((m >> (L2B - 1 - k)) & 1) acc += w[k];
    Y[i] = acc;
}

// ---------------------------------------------------------------------------
// BatchNorm stats over 4096 rows (atomic column reduce), then apply+ReLU+cvt
// ---------------------------------------------------------------------------
__global__ void bn_stats_kernel(const float* __restrict__ Y, float* __restrict__ sum,
                                float* __restrict__ sumsq) {
    int c  = blockIdx.x * blockDim.x + threadIdx.x;  // column
    int r0 = blockIdx.y * 256;                       // row chunk
    float s = 0.f, ss = 0.f;
    for (int r = 0; r < 256; ++r) {
        float v = Y[(size_t)(r0 + r) * HID + c];
        s += v; ss += v * v;
    }
    atomicAdd(&sum[c], s);
    atomicAdd(&sumsq[c], ss);
}

__global__ void bn_apply_kernel(const float* __restrict__ Y, const float* __restrict__ sum,
                                const float* __restrict__ sumsq, const float* __restrict__ gamma,
                                const float* __restrict__ beta, _Float16* __restrict__ X) {
    int i = blockIdx.x * blockDim.x + threadIdx.x;   // over KCB*HID
    int c = i & (HID - 1);
    const float inv_n = 1.f / (float)KCB;
    float m   = sum[c] * inv_n;
    float var = sumsq[c] * inv_n - m * m;
    float sc  = rsqrtf(var + 1e-5f) * gamma[c];
    float v   = (Y[i] - m) * sc + beta[c];
    X[i] = (_Float16)fmaxf(v, 0.f);
}

// ---------------------------------------------------------------------------
// Mid GEMM: Y[4096,1024] = X_f16[4096,1024] @ W_f16[1024,1024]^T + b
// Block = 8 waves; each wave owns one 16-row strip x 4 N-tiles (16x64 out).
// All waves share one 64-column weight tile, staged through LDS by the
// Tensor Data Mover (double-buffered K-chunks of 128 halves), synced with
// s_wait_tensorcnt + workgroup barrier.  Fallback: direct global B loads.
// Grid: 32 m-groups x 16 n-groups = 512 blocks of 256 threads.
// ---------------------------------------------------------------------------
__global__ void __launch_bounds__(256) gemm_mid_kernel(const _Float16* __restrict__ X,
                                                       const _Float16* __restrict__ W,
                                                       const float* __restrict__ bias,
                                                       float* __restrict__ Y) {
    int widx = threadIdx.x >> 5;
    int lane = threadIdx.x & 31, lo = lane & 15, hi = lane >> 4;
    int mg = blockIdx.x >> 4;                 // 0..31  (8 m-tiles each)
    int ng = blockIdx.x & 15;                 // 0..15  (4 n-tiles each)
    int mt = mg * 8 + widx;
    int n0 = ng * 64;

    const _Float16* Xrow = X + (size_t)(mt * 16 + lo) * HID;
    v8f acc[4] = {};

#if HAVE_TDM
    __shared__ _Float16 Bs[2][64 * KC];       // 2 x 16 KB double buffer
    const _Float16* Wt = W + (size_t)n0 * HID;

    if (widx == 0) tdm_load_tile(Wt, (unsigned)(unsigned long long)&Bs[0][0]);

    for (int kc = 0; kc < HID; kc += KC) {
        int buf = (kc / KC) & 1;
        if (kc + KC < HID) {
            if (widx == 0) {
                tdm_load_tile(Wt + (kc + KC), (unsigned)(unsigned long long)&Bs[buf ^ 1][0]);
                __builtin_amdgcn_s_wait_tensorcnt(1);   // current chunk landed
            }
        } else {
            if (widx == 0) __builtin_amdgcn_s_wait_tensorcnt(0);
        }
        __syncthreads();                       // LDS chunk visible to all waves
        const _Float16* Bbase = &Bs[buf][0];
#pragma unroll
        for (int ks = 0; ks < KC; ks += 32) {
            Frag16 A;
            A.p[0] = *(const h8*)(Xrow + kc + ks + 16 * hi);
            A.p[1] = *(const h8*)(Xrow + kc + ks + 16 * hi + 8);
#pragma unroll
            for (int ntl = 0; ntl < 4; ++ntl) {
                Frag16 Bf;
                const _Float16* Br = Bbase + (size_t)(ntl * 16 + lo) * KC + ks + 16 * hi;
                Bf.p[0] = *(const h8*)(Br);
                Bf.p[1] = *(const h8*)(Br + 8);
                acc[ntl] = wmma_f16(A.v, Bf.v, acc[ntl]);
            }
        }
        __syncthreads();                       // all reads done before buf reuse
    }
#else
    for (int k = 0; k < HID; k += 32) {
        Frag16 A;
        A.p[0] = *(const h8*)(Xrow + k + 16 * hi);
        A.p[1] = *(const h8*)(Xrow + k + 16 * hi + 8);
#pragma unroll
        for (int ntl = 0; ntl < 4; ++ntl) {
            const _Float16* Wr = W + (size_t)(n0 + ntl * 16 + lo) * HID + 16 * hi;
            Frag16 Bf;
            Bf.p[0] = *(const h8*)(Wr + k);
            Bf.p[1] = *(const h8*)(Wr + k + 8);
            acc[ntl] = wmma_f16(A.v, Bf.v, acc[ntl]);
        }
    }
#endif

    FragC C;
#pragma unroll
    for (int ntl = 0; ntl < 4; ++ntl) {
        C.v = acc[ntl];
        float b = bias[n0 + ntl * 16 + lo];
#pragma unroll
        for (int r = 0; r < 8; ++r)
            Y[(size_t)(mt * 16 + 8 * hi + r) * HID + n0 + ntl * 16 + lo] = C.f[r] + b;
    }
}

// ---------------------------------------------------------------------------
// Final codebook layer: embed = l2norm(X_f16 @ Wout^T + b_out)
// wave owns 16 rows x all 4 N-tiles; cross-lane butterfly for row L2 norm.
// Also emits enorm[m] = ||embed_row||^2 (post-normalization).
// ---------------------------------------------------------------------------
__global__ void embed_kernel(const _Float16* __restrict__ X, const _Float16* __restrict__ Wout,
                             const float* __restrict__ bout, _Float16* __restrict__ E,
                             float* __restrict__ enorm) {
    int wid  = (blockIdx.x * blockDim.x + threadIdx.x) >> 5;   // 0..255 (row chunk)
    int lane = threadIdx.x & 31, lo = lane & 15, hi = lane >> 4;

    const _Float16* Xrow = X + (size_t)(wid * 16 + lo) * HID;
    v8f acc[4] = {};
    for (int k = 0; k < HID; k += 32) {
        Frag16 A;
        A.p[0] = *(const h8*)(Xrow + k + 16 * hi);
        A.p[1] = *(const h8*)(Xrow + k + 16 * hi + 8);
#pragma unroll
        for (int nt = 0; nt < 4; ++nt) {
            Frag16 Bf;
            const _Float16* Wr = Wout + (size_t)(nt * 16 + lo) * HID + 16 * hi + k;
            Bf.p[0] = *(const h8*)(Wr);
            Bf.p[1] = *(const h8*)(Wr + 8);
            acc[nt] = wmma_f16(A.v, Bf.v, acc[nt]);
        }
    }
    FragC C[4];
    float ssq[8];
#pragma unroll
    for (int r = 0; r < 8; ++r) ssq[r] = 0.f;
#pragma unroll
    for (int nt = 0; nt < 4; ++nt) {
        C[nt].v = acc[nt];
        float b = bout[nt * 16 + lo];
#pragma unroll
        for (int r = 0; r < 8; ++r) { C[nt].f[r] += b; ssq[r] += C[nt].f[r] * C[nt].f[r]; }
    }
    // reduce across the 16 lanes of each half-wave (columns of this row)
#pragma unroll
    for (int msk = 1; msk < 16; msk <<= 1)
#pragma unroll
        for (int r = 0; r < 8; ++r) ssq[r] += __shfl_xor(ssq[r], msk, 32);

#pragma unroll
    for (int r = 0; r < 8; ++r) {
        float nrm = sqrtf(ssq[r]);
        float cn  = 1.f / (nrm + 1e-6f);
        int   m   = wid * 16 + 8 * hi + r;
#pragma unroll
        for (int nt = 0; nt < 4; ++nt)
            E[(size_t)m * VQ + nt * 16 + lo] = (_Float16)(C[nt].f[r] * cn);
        if (lo == 0) { float q = nrm * cn; enorm[m] = q * q; }
    }
}

// ---------------------------------------------------------------------------
// Projection: S[v,t] = proj_w[v,:] . h_in[b,:,t] (+ proj_b), then row(t) L2
// norm; wave owns 16 t-columns x all 4 v-tiles. B frag built by f32->f16 cvt.
// Output hn_f16[b*T+t][v], one packed 16B store per (lane, v-tile).
// ---------------------------------------------------------------------------
__global__ void proj_kernel(const float* __restrict__ Hin, const _Float16* __restrict__ Wp,
                            const float* __restrict__ pb, _Float16* __restrict__ HN) {
    int wid  = (blockIdx.x * blockDim.x + threadIdx.x) >> 5;   // 0..1023
    int b = wid >> 7, tb = wid & 127;                          // T/16 == 128
    int lane = threadIdx.x & 31, lo = lane & 15, hi = lane >> 4;

    const float* Hb = Hin + (size_t)b * DD * TT;
    int t = tb * 16 + lo;

    v8f acc[4] = {};
    for (int k = 0; k < DD; k += 32) {
        Frag16 Bf;               // B element (kk, n=t) = h_in[b][kk][t]
#pragma unroll
        for (int j = 0; j < 16; ++j)
            Bf.h[j] = (_Float16)Hb[(size_t)(k + 16 * hi + j) * TT + t];
#pragma unroll
        for (int mt = 0; mt < 4; ++mt) {
            Frag16 A;
            const _Float16* Wr = Wp + (size_t)(mt * 16 + lo) * DD;
            A.p[0] = *(const h8*)(Wr + k + 16 * hi);
            A.p[1] = *(const h8*)(Wr + k + 16 * hi + 8);
            acc[mt] = wmma_f16(A.v, Bf.v, acc[mt]);
        }
    }
    FragC C[4];
    float ssq = 0.f;
#pragma unroll
    for (int mt = 0; mt < 4; ++mt) {
        C[mt].v = acc[mt];
#pragma unroll
        for (int r = 0; r < 8; ++r) {
            C[mt].f[r] += pb[16 * mt + 8 * hi + r];
            ssq += C[mt].f[r] * C[mt].f[r];
        }
    }
    ssq += __shfl_xor(ssq, 16, 32);   // lanes l and l^16 hold the same column t
    float cn = 1.f / (sqrtf(ssq) + 1e-6f);

    size_t row = (size_t)b * TT + t;
#pragma unroll
    for (int mt = 0; mt < 4; ++mt) {
        Frag16 o;
#pragma unroll
        for (int r = 0; r < 8; ++r) o.h[r] = (_Float16)(C[mt].f[r] * cn);
        *(h8*)(HN + row * VQ + 16 * mt + 8 * hi) = o.p[0];
    }
}

// ---------------------------------------------------------------------------
// Fused distance + argmax: score = 2*h.e - ||e||^2 ; wave owns 16 rows,
// loops all 256 codebook tiles, keeps running per-lane best, butterfly argmax.
// ---------------------------------------------------------------------------
__global__ void argmax_kernel(const _Float16* __restrict__ HN, const _Float16* __restrict__ E,
                              const float* __restrict__ enorm, const int* __restrict__ amask,
                              int* __restrict__ code, int* __restrict__ vq_out) {
    int wid  = (blockIdx.x * blockDim.x + threadIdx.x) >> 5;   // 0..1023
    int lane = threadIdx.x & 31, lo = lane & 15, hi = lane >> 4;

    const _Float16* Hrow = HN + (size_t)(wid * 16 + lo) * VQ;
    Frag16 A0, A1;
    A0.p[0] = *(const h8*)(Hrow + 16 * hi);
    A0.p[1] = *(const h8*)(Hrow + 16 * hi + 8);
    A1.p[0] = *(const h8*)(Hrow + 32 + 16 * hi);
    A1.p[1] = *(const h8*)(Hrow + 32 + 16 * hi + 8);

    float best[8]; int bidx[8];
#pragma unroll
    for (int r = 0; r < 8; ++r) { best[r] = -3.0e38f; bidx[r] = 0; }

    for (int nt = 0; nt < KCB / 16; ++nt) {
        const _Float16* Er = E + (size_t)(nt * 16 + lo) * VQ + 16 * hi;
        Frag16 B0, B1;
        B0.p[0] = *(const h8*)(Er);
        B0.p[1] = *(const h8*)(Er + 8);
        B1.p[0] = *(const h8*)(Er + 32);
        B1.p[1] = *(const h8*)(Er + 40);
        v8f acc = {};
        acc = wmma_f16(A0.v, B0.v, acc);
        acc = wmma_f16(A1.v, B1.v, acc);
        FragC C; C.v = acc;
        int   n  = nt * 16 + lo;
        float en = enorm[n];
#pragma unroll
        for (int r = 0; r < 8; ++r) {
            float s = 2.f * C.f[r] - en;
            if (s > best[r]) { best[r] = s; bidx[r] = n; }
        }
    }
    // butterfly argmax over the 16 columns held by each half-wave
#pragma unroll
    for (int msk = 1; msk < 16; msk <<= 1) {
#pragma unroll
        for (int r = 0; r < 8; ++r) {
            float os = __shfl_xor(best[r], msk, 32);
            int   oi = __shfl_xor(bidx[r], msk, 32);
            if (os > best[r] || (os == best[r] && oi < bidx[r])) { best[r] = os; bidx[r] = oi; }
        }
    }
    if (lo == 0) {
#pragma unroll
        for (int r = 0; r < 8; ++r) {
            int row = wid * 16 + 8 * hi + r;
            code[row]   = bidx[r];
            vq_out[row] = amask[row] ? bidx[r] : 0;
        }
    }
}

// ---------------------------------------------------------------------------
// Output: quantized[m,d] = (mask[m] ? embed[code[m]] : 0) @ inv_w^T + inv_b
// A fragments gathered per-lane from embed_f16 rows selected by code.
// ---------------------------------------------------------------------------
__global__ void outproj_kernel(const _Float16* __restrict__ E, const int* __restrict__ code,
                               const int* __restrict__ amask, const _Float16* __restrict__ Winv,
                               const float* __restrict__ binv, float* __restrict__ Q) {
    int wid  = (blockIdx.x * blockDim.x + threadIdx.x) >> 5;
    int lane = threadIdx.x & 31, lo = lane & 15, hi = lane >> 4;
    const int ntiles = DD / 16;                    // 48
    int mt = wid / ntiles, nt = wid % ntiles;

    int m  = mt * 16 + lo;
    Frag16 A0 = {}, A1 = {};
    if (amask[m]) {
        const _Float16* Erow = E + (size_t)code[m] * VQ;
        A0.p[0] = *(const h8*)(Erow + 16 * hi);
        A0.p[1] = *(const h8*)(Erow + 16 * hi + 8);
        A1.p[0] = *(const h8*)(Erow + 32 + 16 * hi);
        A1.p[1] = *(const h8*)(Erow + 32 + 16 * hi + 8);
    }
    const _Float16* Wr = Winv + (size_t)(nt * 16 + lo) * VQ + 16 * hi;
    Frag16 B0, B1;
    B0.p[0] = *(const h8*)(Wr);
    B0.p[1] = *(const h8*)(Wr + 8);
    B1.p[0] = *(const h8*)(Wr + 32);
    B1.p[1] = *(const h8*)(Wr + 40);

    v8f acc = {};
    acc = wmma_f16(A0.v, B0.v, acc);
    acc = wmma_f16(A1.v, B1.v, acc);
    FragC C; C.v = acc;
    float b = binv[nt * 16 + lo];
#pragma unroll
    for (int r = 0; r < 8; ++r)
        Q[(size_t)(mt * 16 + 8 * hi + r) * DD + nt * 16 + lo] = C.f[r] + b;
}

// ---------------------------------------------------------------------------
// Host-side orchestration (graph-capture safe: only async launches/memsets)
// ---------------------------------------------------------------------------
extern "C" void kernel_launch(void* const* d_in, const int* in_sizes, int n_in,
                              void* d_out, int out_size, void* d_ws, size_t ws_size,
                              hipStream_t stream) {
    (void)in_sizes; (void)n_in; (void)out_size; (void)ws_size;

    const float* h_in     = (const float*)d_in[0];
    const int*   amask    = (const int*)  d_in[1];
    const float* proj_w   = (const float*)d_in[2];
    const float* proj_b   = (const float*)d_in[3];
    const float* inv_w    = (const float*)d_in[4];
    const float* inv_b    = (const float*)d_in[5];
    const float* mlp_w_in = (const float*)d_in[6];
    const float* mlp_b_in = (const float*)d_in[7];
    const float* w_mid    = (const float*)d_in[8];
    const float* b_mid    = (const float*)d_in[9];
    const float* w_out    = (const float*)d_in[10];
    const float* b_out    = (const float*)d_in[11];
    const float* bn_gamma = (const float*)d_in[12];
    const float* bn_beta  = (const float*)d_in[13];

    float* quantized = (float*)d_out;                       // [B,T,D] f32
    int*   vq_code   = (int*)d_out + (size_t)BB * TT * DD;  // [B,T] i32

    // workspace layout (256B aligned slots)
    char* ws = (char*)d_ws;
    size_t off = 0;
    auto take = [&](size_t bytes) { char* p = ws + off; off += (bytes + 255) & ~(size_t)255; return p; };
    _Float16* Wmid16 = (_Float16*)take((size_t)NMID * HID * HID * 2);
    _Float16* Wout16 = (_Float16*)take((size_t)VQ * HID * 2);
    _Float16* Wprj16 = (_Float16*)take((size_t)VQ * DD * 2);
    _Float16* Winv16 = (_Float16*)take((size_t)DD * VQ * 2);
    float*    Ybuf   = (float*)   take((size_t)KCB * HID * 4);
    _Float16* Xbuf   = (_Float16*)take((size_t)KCB * HID * 2);
    _Float16* E16    = (_Float16*)take((size_t)KCB * VQ * 2);
    float*    enorm  = (float*)   take((size_t)KCB * 4);
    _Float16* HN16   = (_Float16*)take((size_t)ROWS_BT * VQ * 2);
    int*      codeb  = (int*)     take((size_t)ROWS_BT * 4);
    float*    stats  = (float*)   take((size_t)2 * HID * 4);   // sum | sumsq

    // --- convert weights to f16 once per call ---
    {
        int n = NMID * HID * HID;
        cvt_f16_kernel<<<(n + 255) / 256, 256, 0, stream>>>(w_mid, Wmid16, n);
        n = VQ * HID;
        cvt_f16_kernel<<<(n + 255) / 256, 256, 0, stream>>>(w_out, Wout16, n);
        n = VQ * DD;
        cvt_f16_kernel<<<(n + 255) / 256, 256, 0, stream>>>(proj_w, Wprj16, n);
        n = DD * VQ;
        cvt_f16_kernel<<<(n + 255) / 256, 256, 0, stream>>>(inv_w, Winv16, n);
    }

    // --- codebook MLP ---
    codebook_l0_kernel<<<(KCB * HID) / 256, 256, 0, stream>>>(mlp_w_in, mlp_b_in, Ybuf);

    for (int layer = 0; layer <= NMID; ++layer) {
        hipMemsetAsync(stats, 0, 2 * HID * sizeof(float), stream);
        bn_stats_kernel<<<dim3(HID / 256, KCB / 256), 256, 0, stream>>>(Ybuf, stats, stats + HID);
        bn_apply_kernel<<<(KCB * HID) / 256, 256, 0, stream>>>(
            Ybuf, stats, stats + HID, bn_gamma + layer * HID, bn_beta + layer * HID, Xbuf);
        if (layer < NMID) {
            // 32 m-groups x 16 n-groups, 8 waves/block
            gemm_mid_kernel<<<512, 256, 0, stream>>>(
                Xbuf, Wmid16 + (size_t)layer * HID * HID, b_mid + layer * HID, Ybuf);
        }
    }
    embed_kernel<<<(KCB / 16) * 32 / 256, 256, 0, stream>>>(Xbuf, Wout16, b_out, E16, enorm);

    // --- hidden projection + l2norm ---
    proj_kernel<<<(BB * TT / 16) * 32 / 256, 256, 0, stream>>>(h_in, Wprj16, proj_b, HN16);

    // --- fused distance + argmax + vq_code ---
    argmax_kernel<<<(ROWS_BT / 16) * 32 / 256, 256, 0, stream>>>(
        HN16, E16, enorm, amask, codeb, vq_code);

    // --- gather + inverse projection ---
    {
        int waves = (ROWS_BT / 16) * (DD / 16);      // 49152
        outproj_kernel<<<waves * 32 / 256, 256, 0, stream>>>(
            E16, codeb, amask, Winv16, inv_b, quantized);
    }
}
